// ModelMRM_73675868996327
// MI455X (gfx1250) — matrix operand order repaired and checked by reading, not compiled, not run on hardware
//
#include <hip/hip_runtime.h>
#include <hip/hip_fp16.h>

typedef _Float16 half_t;
typedef __attribute__((ext_vector_type(16))) _Float16 v16h;
typedef __attribute__((ext_vector_type(8)))  float    v8f;

#define B_    64
#define L_    256
#define NSEQ  (B_ * L_)
#define P_    32
#define E_    128
#define DEPTH_ 4
#define HEADS_ 8
#define HD_   16
#define MLP_  512

// ---- f16 packed-weight layout in d_ws (units: half elements) ----
#define OFF_WQKV 0
#define SZ_WQKV  (DEPTH_ * E_ * 3 * E_)     // 196608
#define OFF_WO   (OFF_WQKV + SZ_WQKV)
#define SZ_WO    (DEPTH_ * E_ * E_)         // 65536
#define OFF_WM1  (OFF_WO + SZ_WO)
#define SZ_WM1   (DEPTH_ * E_ * MLP_)       // 262144
#define OFF_WM2  (OFF_WM1 + SZ_WM1)
#define SZ_WM2   (DEPTH_ * MLP_ * E_)       // 262144
#define OFF_WD1  (OFF_WM2 + SZ_WM2)
#define SZ_WD1   (E_ * MLP_)                // 65536
#define WS_HALVES (OFF_WD1 + SZ_WD1)
#define ACC_BYTE_OFF (WS_HALVES * 2)        // then 2 floats: num, cnt

// ---------------- WMMA fragment helpers (contiguous k = 16*(lane>>4) + s order) ----------------
// A fragment: 16x32 f16, M = lane&15, K base = (lane>>4)*16.
template <int KMAX>
__device__ __forceinline__ v16h ldA(const half_t* base, int rs, int cs, int lane) {
  const int m = lane & 15, hh = lane >> 4;
  v16h a;
#pragma unroll
  for (int v = 0; v < 8; ++v) {
    const int k0 = 2 * v + 16 * hh;
    a[2 * v]     = (k0     < KMAX) ? base[m * rs + (k0)     * cs] : (half_t)0.f;
    a[2 * v + 1] = (k0 + 1 < KMAX) ? base[m * rs + (k0 + 1) * cs] : (half_t)0.f;
  }
  return a;
}

// B fragment: 32x16 f16, N = lane&15, K base = (lane>>4)*16.
template <int KMAX>
__device__ __forceinline__ v16h ldB(const half_t* base, int rs, int cs, int lane) {
  const int n = lane & 15, kb = (lane >> 4) * 16;
  v16h b;
#pragma unroll
  for (int v = 0; v < 8; ++v) {
    const int k0 = kb + 2 * v;
    b[2 * v]     = (k0     < KMAX) ? base[(k0)     * rs + n * cs] : (half_t)0.f;
    b[2 * v + 1] = (k0 + 1 < KMAX) ? base[(k0 + 1) * rs + n * cs] : (half_t)0.f;
  }
  return b;
}

// Pre-packed B fragment: 32B per lane, contiguous & coalesced, straight from L2.
__device__ __forceinline__ v16h ldBpk(const half_t* wb, int KT, int kt, int nt, int lane) {
  return *((const v16h*)wb + ((nt * KT + kt) * 32 + lane));
}

__device__ __forceinline__ v8f wmma16(v16h a, v16h b, v8f c) {
  return __builtin_amdgcn_wmma_f32_16x16x32_f16(false, a, false, b, (short)0, c,
                                                false, false);
}

__device__ __forceinline__ float gelu_f(float x) {
  return 0.5f * x * (1.f + erff(x * 0.70710678118654752f));
}

// ---------------- shared memory ----------------
struct Smem {
  alignas(16) float  xf[P_ * E_];            // 16 KB residual stream (f32)
  alignas(16) half_t hf[P_ * E_];            // 8 KB  LN output / attn-O (f16)
  alignas(16) half_t qkv[P_ * 3 * E_];       // 24 KB q|k|v (f16)
  alignas(16) float  scr[P_ * 256];          // 32 KB scores f32  OR  MLP hidden f16
  alignas(16) half_t attf[HEADS_ * P_ * P_]; // 16 KB softmaxed probs (f16)
  float pe1[E_], pe[E_];
  float red1[256], red2[256];
  float mu[P_], rst[P_];
  float lnorm[P_ * 2];
  int   selp[P_];
  float s_num, s_cnt;
};

__device__ void layernorm(const float* xf, half_t* dst, const float* s,
                          const float* b, Smem& sm) {
  const int tid = threadIdx.x;
  const int p = tid >> 3, j = tid & 7;
  float s1 = 0.f, s2 = 0.f;
#pragma unroll
  for (int ii = 0; ii < 16; ++ii) {
    const float v = xf[p * E_ + j * 16 + ii];
    s1 += v; s2 += v * v;
  }
  sm.red1[tid] = s1; sm.red2[tid] = s2;
  __syncthreads();
  if (j == 0) {
    float a = 0.f, c = 0.f;
#pragma unroll
    for (int q = 0; q < 8; ++q) { a += sm.red1[p * 8 + q]; c += sm.red2[p * 8 + q]; }
    const float m = a * (1.f / E_);
    const float var = c * (1.f / E_) - m * m;
    sm.mu[p] = m; sm.rst[p] = rsqrtf(var + 1e-5f);
  }
  __syncthreads();
  const float m = sm.mu[p], r = sm.rst[p];
#pragma unroll
  for (int ii = 0; ii < 16; ++ii) {
    const int e = j * 16 + ii;
    dst[p * E_ + e] = (half_t)((xf[p * E_ + e] - m) * r * s[e] + b[e]);
  }
  __syncthreads();
}

// ---------------- kernels ----------------
__global__ void init_acc_kernel(float* acc) { acc[0] = 0.f; acc[1] = 0.f; }

// Pack f32 row-major (K,N) weights into WMMA B-fragment order, f16.
__global__ void pack_b_kernel(const float* __restrict__ src, half_t* __restrict__ dst,
                              int K, int N, int count) {
  const int KT = K >> 5;
  const long total = (long)count * K * N;
  for (long idx = (long)blockIdx.x * blockDim.x + threadIdx.x; idx < total;
       idx += (long)gridDim.x * blockDim.x) {
    const long mat = idx / ((long)K * N);
    const int  rem = (int)(idx - mat * K * N);
    const int  s = rem & 15;
    const int  lane2 = (rem >> 4) & 31;
    const int  tile = rem >> 9;
    const int  kt = tile % KT, nt = tile / KT;
    const int  k = kt * 32 + ((lane2 >> 4) << 4) + s;
    const int  n = nt * 16 + (lane2 & 15);
    dst[idx] = (half_t)src[mat * K * N + (long)k * N + n];
  }
}

__global__ void finalize_kernel(const float* acc, float* out) {
  out[0] = acc[0] / fmaxf(acc[1] * 2.0f, 1.0f);
}

__global__ __launch_bounds__(256)
void mrm_transformer(const float* __restrict__ pos, const float* __restrict__ centers,
                     const float* __restrict__ angles, const unsigned char* __restrict__ sel,
                     const float* __restrict__ Wp, const float* __restrict__ bp,
                     const float* __restrict__ Wpe1, const float* __restrict__ bpe1,
                     const float* __restrict__ Wpe2, const float* __restrict__ bpe2,
                     const float* __restrict__ lane_type, const float* __restrict__ mask_token,
                     const float* __restrict__ bqkv, const float* __restrict__ bo,
                     const float* __restrict__ ln1_s, const float* __restrict__ ln1_b,
                     const float* __restrict__ bm1, const float* __restrict__ bm2,
                     const float* __restrict__ ln2_s, const float* __restrict__ ln2_b,
                     const float* __restrict__ lnf_s, const float* __restrict__ lnf_b,
                     const float* __restrict__ bd1, const float* __restrict__ Wd2,
                     const float* __restrict__ bd2, const half_t* __restrict__ wsh,
                     float* __restrict__ y_out, float* __restrict__ acc) {
  __shared__ Smem sm;
  const int tid = threadIdx.x;
  const int wave = tid >> 5;
  const int lane = tid & 31;
  const int seq = blockIdx.x;

  // ---------------- embedding ----------------
  const float cx = centers[seq * 2], cy = centers[seq * 2 + 1];
  const float ang = angles[seq];
  if (tid < E_) {
    const float c = cosf(ang), snn = sinf(ang);
    float v = cx * Wpe1[tid] + cy * Wpe1[E_ + tid] + c * Wpe1[2 * E_ + tid] +
              snn * Wpe1[3 * E_ + tid] + bpe1[tid];
    sm.pe1[tid] = gelu_f(v);
  }
  if (tid < P_) {
    const float px = pos[(seq * P_ + tid) * 2], py = pos[(seq * P_ + tid) * 2 + 1];
    sm.lnorm[tid * 2] = px - cx; sm.lnorm[tid * 2 + 1] = py - cy;
    sm.selp[tid] = (int)sel[seq * P_ + tid];
  }
  if (tid == 0) { sm.s_num = 0.f; sm.s_cnt = 0.f; }
  __syncthreads();
  if (tid < E_) {
    float v = bpe2[tid];
#pragma unroll 4
    for (int k = 0; k < E_; ++k) v += sm.pe1[k] * Wpe2[k * E_ + tid];
    sm.pe[tid] = v;
  }
  __syncthreads();
  for (int idx = tid; idx < P_ * E_; idx += 256) {
    const int p = idx >> 7, e = idx & 127;
    float v;
    if (sm.selp[p]) v = mask_token[e];
    else v = sm.lnorm[2 * p] * Wp[e] + sm.lnorm[2 * p + 1] * Wp[E_ + e] + bp[e] +
             sm.pe[e] + lane_type[e];
    sm.xf[idx] = v;
  }
  __syncthreads();

  half_t* mlph = (half_t*)(void*)sm.scr;  // 32x512 f16 hidden, aliases score buf
  float*  att32 = sm.scr;                 // 8x32x32 f32 scores

  // ---------------- transformer layers ----------------
  for (int layer = 0; layer < DEPTH_; ++layer) {
    const half_t* wqkv = wsh + OFF_WQKV + layer * E_ * 3 * E_;
    const half_t* wo   = wsh + OFF_WO   + layer * E_ * E_;
    const half_t* wm1  = wsh + OFF_WM1  + layer * E_ * MLP_;
    const half_t* wm2  = wsh + OFF_WM2  + layer * MLP_ * E_;

    layernorm(sm.xf, sm.hf, ln1_s + layer * E_, ln1_b + layer * E_, sm);

    // ---- QKV: (32x128)@(128x384), B streamed packed from L2 ----
    for (int nt = wave; nt < 24; nt += 8) {
      v8f acc0 = {}, acc1 = {};
#pragma unroll
      for (int kt = 0; kt < 4; ++kt) {
        const v16h bf = ldBpk(wqkv, 4, kt, nt, lane);
        const v16h a0 = ldA<32>(sm.hf + 0 * E_ + kt * 32, E_, 1, lane);
        const v16h a1 = ldA<32>(sm.hf + 16 * E_ + kt * 32, E_, 1, lane);
        acc0 = wmma16(a0, bf, acc0);
        acc1 = wmma16(a1, bf, acc1);
      }
      const int n = nt * 16 + (lane & 15);
      const int mb = (lane >> 4) << 3;
      const float bias = bqkv[layer * 3 * E_ + n];
#pragma unroll
      for (int r = 0; r < 8; ++r) {
        sm.qkv[(mb + r) * (3 * E_) + n]      = (half_t)(acc0[r] + bias);
        sm.qkv[(16 + mb + r) * (3 * E_) + n] = (half_t)(acc1[r] + bias);
      }
    }
    __syncthreads();

    // warm L2/L0 for O-projection weights while attention runs
    {
      const char* pf = (const char*)wo;
      for (int o = tid; o < (E_ * E_ * 2) / 128; o += 256)
        __builtin_prefetch(pf + o * 128, 0, 1);
    }

    // ---- S = Q K^T (HD=16 zero-padded to K=32) ----
    for (int t = wave; t < 32; t += 8) {
      const int h = t >> 2, mt = (t >> 1) & 1, nt = t & 1;
      const v16h a = ldA<16>(sm.qkv + (mt * 16) * (3 * E_) + h * HD_, 3 * E_, 1, lane);
      const v16h bf = ldB<16>(sm.qkv + (nt * 16) * (3 * E_) + E_ + h * HD_, 1, 3 * E_, lane);
      v8f accs = {};
      accs = wmma16(a, bf, accs);
      const int n = nt * 16 + (lane & 15);
      const int mb = mt * 16 + ((lane >> 4) << 3);
#pragma unroll
      for (int r = 0; r < 8; ++r)
        att32[h * (P_ * P_) + (mb + r) * P_ + n] = accs[r];
    }
    __syncthreads();

    // ---- softmax (one row per thread: 8 heads x 32 rows) ----
    {
      const int h = tid >> 5, r = tid & 31;
      const float scale = 0.25f;  // 1/sqrt(16)
      float mx = -1e30f;
      for (int c = 0; c < P_; ++c)
        mx = fmaxf(mx, att32[h * 1024 + r * 32 + c] * scale);
      float ssum = 0.f;
      for (int c = 0; c < P_; ++c) {
        const float e = expf(att32[h * 1024 + r * 32 + c] * scale - mx);
        ssum += e;
        sm.attf[h * 1024 + r * 32 + c] = (half_t)e;
      }
      const float inv = 1.f / ssum;
      for (int c = 0; c < P_; ++c)
        sm.attf[h * 1024 + r * 32 + c] =
            (half_t)((float)sm.attf[h * 1024 + r * 32 + c] * inv);
    }
    __syncthreads();

    // ---- O = att @ V : per head, K=32 exact, output -> hf ----
    for (int h = wave; h < HEADS_; h += 8) {
      const v16h bf = ldB<32>(sm.qkv + 2 * E_ + h * HD_, 3 * E_, 1, lane);
      const v16h a0 = ldA<32>(sm.attf + h * 1024 + 0 * 32, 32, 1, lane);
      const v16h a1 = ldA<32>(sm.attf + h * 1024 + 16 * 32, 32, 1, lane);
      v8f acc0 = {}, acc1 = {};
      acc0 = wmma16(a0, bf, acc0);
      acc1 = wmma16(a1, bf, acc1);
      const int n = lane & 15;
      const int mb = (lane >> 4) << 3;
#pragma unroll
      for (int r = 0; r < 8; ++r) {
        sm.hf[(mb + r) * E_ + h * HD_ + n]      = (half_t)acc0[r];
        sm.hf[(16 + mb + r) * E_ + h * HD_ + n] = (half_t)acc1[r];
      }
    }
    __syncthreads();

    // ---- x += O @ Wo + bo ----
    for (int nt = wave; nt < 8; nt += 8) {
      v8f acc0 = {}, acc1 = {};
#pragma unroll
      for (int kt = 0; kt < 4; ++kt) {
        const v16h bf = ldBpk(wo, 4, kt, nt, lane);
        const v16h a0 = ldA<32>(sm.hf + 0 * E_ + kt * 32, E_, 1, lane);
        const v16h a1 = ldA<32>(sm.hf + 16 * E_ + kt * 32, E_, 1, lane);
        acc0 = wmma16(a0, bf, acc0);
        acc1 = wmma16(a1, bf, acc1);
      }
      const int n = nt * 16 + (lane & 15);
      const int mb = (lane >> 4) << 3;
      const float bias = bo[layer * E_ + n];
#pragma unroll
      for (int r = 0; r < 8; ++r) {
        sm.xf[(mb + r) * E_ + n]      += acc0[r] + bias;
        sm.xf[(16 + mb + r) * E_ + n] += acc1[r] + bias;
      }
    }
    __syncthreads();

    layernorm(sm.xf, sm.hf, ln2_s + layer * E_, ln2_b + layer * E_, sm);

    // ---- hidden = gelu(h @ Wm1 + bm1) -> mlph (f16) ----
    for (int nt = wave; nt < 32; nt += 8) {
      v8f acc0 = {}, acc1 = {};
#pragma unroll
      for (int kt = 0; kt < 4; ++kt) {
        const v16h bf = ldBpk(wm1, 4, kt, nt, lane);
        const v16h a0 = ldA<32>(sm.hf + 0 * E_ + kt * 32, E_, 1, lane);
        const v16h a1 = ldA<32>(sm.hf + 16 * E_ + kt * 32, E_, 1, lane);
        acc0 = wmma16(a0, bf, acc0);
        acc1 = wmma16(a1, bf, acc1);
      }
      const int n = nt * 16 + (lane & 15);
      const int mb = (lane >> 4) << 3;
      const float bias = bm1[layer * MLP_ + n];
#pragma unroll
      for (int r = 0; r < 8; ++r) {
        mlph[(mb + r) * MLP_ + n]      = (half_t)gelu_f(acc0[r] + bias);
        mlph[(16 + mb + r) * MLP_ + n] = (half_t)gelu_f(acc1[r] + bias);
      }
    }
    __syncthreads();

    // ---- x += hidden @ Wm2 + bm2 (K=512) ----
    for (int nt = wave; nt < 8; nt += 8) {
      v8f acc0 = {}, acc1 = {};
#pragma unroll 4
      for (int kt = 0; kt < 16; ++kt) {
        const v16h bf = ldBpk(wm2, 16, kt, nt, lane);
        const v16h a0 = ldA<32>(mlph + 0 * MLP_ + kt * 32, MLP_, 1, lane);
        const v16h a1 = ldA<32>(mlph + 16 * MLP_ + kt * 32, MLP_, 1, lane);
        acc0 = wmma16(a0, bf, acc0);
        acc1 = wmma16(a1, bf, acc1);
      }
      const int n = nt * 16 + (lane & 15);
      const int mb = (lane >> 4) << 3;
      const float bias = bm2[layer * E_ + n];
#pragma unroll
      for (int r = 0; r < 8; ++r) {
        sm.xf[(mb + r) * E_ + n]      += acc0[r] + bias;
        sm.xf[(16 + mb + r) * E_ + n] += acc1[r] + bias;
      }
    }
    __syncthreads();
  }

  // ---------------- decoder ----------------
  layernorm(sm.xf, sm.hf, lnf_s, lnf_b, sm);

  const half_t* wd1 = wsh + OFF_WD1;
  for (int nt = wave; nt < 32; nt += 8) {
    v8f acc0 = {}, acc1 = {};
#pragma unroll
    for (int kt = 0; kt < 4; ++kt) {
      const v16h bf = ldBpk(wd1, 4, kt, nt, lane);
      const v16h a0 = ldA<32>(sm.hf + 0 * E_ + kt * 32, E_, 1, lane);
      const v16h a1 = ldA<32>(sm.hf + 16 * E_ + kt * 32, E_, 1, lane);
      acc0 = wmma16(a0, bf, acc0);
      acc1 = wmma16(a1, bf, acc1);
    }
    const int n = nt * 16 + (lane & 15);
    const int mb = (lane >> 4) << 3;
    const float bias = bd1[n];
#pragma unroll
    for (int r = 0; r < 8; ++r) {
      mlph[(mb + r) * MLP_ + n]      = (half_t)fmaxf(acc0[r] + bias, 0.f);
      mlph[(16 + mb + r) * MLP_ + n] = (half_t)fmaxf(acc1[r] + bias, 0.f);
    }
  }
  __syncthreads();

  // y = relu_hidden @ Wd2 + bd2 (N=2, scalar), plus loss partials
  if (tid < 64) {
    const int p = tid >> 1, c = tid & 1;
    float v = bd2[c];
#pragma unroll 8
    for (int k = 0; k < MLP_; ++k) v += (float)mlph[p * MLP_ + k] * Wd2[k * 2 + c];
    y_out[(seq * P_ + p) * 2 + c] = v;
    if (sm.selp[p]) {
      const float d = v - sm.lnorm[p * 2 + c];
      atomicAdd(&sm.s_num, d * d);
    }
  }
  if (tid < P_ && sm.selp[tid]) atomicAdd(&sm.s_cnt, 1.0f);
  __syncthreads();
  if (tid == 0) {
    atomicAdd(&acc[0], sm.s_num);
    atomicAdd(&acc[1], sm.s_cnt);
  }
}

// ---------------- host launch ----------------
extern "C" void kernel_launch(void* const* d_in, const int* in_sizes, int n_in,
                              void* d_out, int out_size, void* d_ws, size_t ws_size,
                              hipStream_t stream) {
  (void)in_sizes; (void)n_in; (void)out_size; (void)ws_size;
  const float* lane_positions = (const float*)d_in[0];
  const float* lane_centers   = (const float*)d_in[1];
  const float* lane_angles    = (const float*)d_in[2];
  const unsigned char* sel_mask = (const unsigned char*)d_in[4];
  const float* Wp = (const float*)d_in[5],  *bp = (const float*)d_in[6];
  const float* Wpe1 = (const float*)d_in[7], *bpe1 = (const float*)d_in[8];
  const float* Wpe2 = (const float*)d_in[9], *bpe2 = (const float*)d_in[10];
  const float* lane_type = (const float*)d_in[11], *mask_token = (const float*)d_in[12];
  const float* Wqkv = (const float*)d_in[13], *bqkv = (const float*)d_in[14];
  const float* Wo = (const float*)d_in[15],   *bo = (const float*)d_in[16];
  const float* ln1_s = (const float*)d_in[17], *ln1_b = (const float*)d_in[18];
  const float* Wm1 = (const float*)d_in[19], *bm1 = (const float*)d_in[20];
  const float* Wm2 = (const float*)d_in[21], *bm2 = (const float*)d_in[22];
  const float* ln2_s = (const float*)d_in[23], *ln2_b = (const float*)d_in[24];
  const float* lnf_s = (const float*)d_in[25], *lnf_b = (const float*)d_in[26];
  const float* Wd1 = (const float*)d_in[27], *bd1 = (const float*)d_in[28];
  const float* Wd2 = (const float*)d_in[29], *bd2 = (const float*)d_in[30];

  half_t* wsh = (half_t*)d_ws;
  float*  accs = (float*)((char*)d_ws + ACC_BYTE_OFF);
  float*  out  = (float*)d_out;

  init_acc_kernel<<<1, 1, 0, stream>>>(accs);
  pack_b_kernel<<<512, 256, 0, stream>>>(Wqkv, wsh + OFF_WQKV, E_, 3 * E_, DEPTH_);
  pack_b_kernel<<<256, 256, 0, stream>>>(Wo,   wsh + OFF_WO,   E_, E_,     DEPTH_);
  pack_b_kernel<<<512, 256, 0, stream>>>(Wm1,  wsh + OFF_WM1,  E_, MLP_,   DEPTH_);
  pack_b_kernel<<<512, 256, 0, stream>>>(Wm2,  wsh + OFF_WM2,  MLP_, E_,   DEPTH_);
  pack_b_kernel<<<256, 256, 0, stream>>>(Wd1,  wsh + OFF_WD1,  E_, MLP_,   1);

  mrm_transformer<<<NSEQ, 256, 0, stream>>>(
      lane_positions, lane_centers, lane_angles, sel_mask, Wp, bp, Wpe1, bpe1,
      Wpe2, bpe2, lane_type, mask_token, bqkv, bo, ln1_s, ln1_b, bm1, bm2,
      ln2_s, ln2_b, lnf_s, lnf_b, bd1, Wd2, bd2, wsh, out + 1, accs);

  finalize_kernel<<<1, 1, 0, stream>>>(accs, out);
}
